// AutoCorrelationLayer_40973988004641
// MI455X (gfx1250) — matrix operand order repaired and checked by reading, not compiled, not run on hardware
//
#include <hip/hip_runtime.h>

typedef __attribute__((ext_vector_type(16))) __bf16 v16bf;
typedef __attribute__((ext_vector_type(8)))  __bf16 v8bf;
typedef __attribute__((ext_vector_type(8)))  float  v8f;

#define WMMA_BF16(a, b, c) \
  __builtin_amdgcn_wmma_f32_16x16x32_bf16(false, (a), false, (b), (short)0, (c), false, false)

static __device__ __forceinline__ v16bf cat16(v8bf lo, v8bf hi) {
  return __builtin_shufflevector(lo, hi, 0, 1, 2, 3, 4, 5, 6, 7,
                                         8, 9, 10, 11, 12, 13, 14, 15);
}

// -----------------------------------------------------------------------------
// x (fp32) -> xb (bf16), elementwise, 8 values per thread, fully coalesced.
// -----------------------------------------------------------------------------
__global__ __launch_bounds__(256) void cvt_bf16_kernel(
    const float* __restrict__ in, __bf16* __restrict__ outb)
{
  const size_t i = (size_t)blockIdx.x * 256 + threadIdx.x;
  const float4* p = (const float4*)in + i * 2;
  float4 a = p[0];
  float4 b = p[1];
  v8bf v;
  v[0] = (__bf16)a.x; v[1] = (__bf16)a.y; v[2] = (__bf16)a.z; v[3] = (__bf16)a.w;
  v[4] = (__bf16)b.x; v[5] = (__bf16)b.y; v[6] = (__bf16)b.z; v[7] = (__bf16)b.w;
  *((v8bf*)outb + i) = v;
}

// -----------------------------------------------------------------------------
// W (fp32, [D][D] k-major) -> Wt (bf16, [D][D] n-major): Wt[n][k] = W[k][n].
// LDS-tiled 64x64 transpose (65-col pad avoids bank conflicts).
// -----------------------------------------------------------------------------
__global__ __launch_bounds__(256) void wtrans_kernel(
    const float* __restrict__ W, __bf16* __restrict__ Wt, int D)
{
  __shared__ float tile[64][65];
  const int n0 = blockIdx.x * 64;
  const int k0 = blockIdx.y * 64;
  const int tx = threadIdx.x & 63;
  const int ty = threadIdx.x >> 6;               // 0..3
  #pragma unroll
  for (int i = 0; i < 16; ++i) {
    const int r = ty + i * 4;
    tile[r][tx] = W[(size_t)(k0 + r) * D + n0 + tx];
  }
  __syncthreads();
  #pragma unroll
  for (int i = 0; i < 16; ++i) {
    const int r = ty + i * 4;                    // output row n = n0 + r
    Wt[(size_t)(n0 + r) * D + k0 + tx] = (__bf16)tile[tx][r];
  }
}

// -----------------------------------------------------------------------------
// QKV projection GEMM, all-bf16 operands, no LDS, no barriers:
//   out = xb[M][D] @ Wt^T + bias      (Wt is n-major: Wt[n][k])
// Block = 256 threads (8 waves) covers a 128x128 output tile for L0/WGP$ reuse;
// wave (wm=w>>1, wn=w&1) computes 2 M-tiles x 4 N-tiles = 8 WMMA accumulators.
// Inner loop per wave per 32-wide k-step: 12 contiguous b128 loads + 8 WMMA.
// -----------------------------------------------------------------------------
__global__ __launch_bounds__(256) void qkv_gemm_kernel(
    const __bf16* __restrict__ xb, const __bf16* __restrict__ Wt,
    const float* __restrict__ bias, __bf16* __restrict__ out,
    int D, int transpose_out)
{
  constexpr int SEQ = 2048;                      // for transposed (V) store

  const int tid  = threadIdx.x;
  const int w    = tid >> 5;
  const int lane = tid & 31;
  const int half = lane >> 4;
  const int ln   = lane & 15;
  const int wm   = w >> 1;                       // 0..3 -> tile rows {2wm,2wm+1}
  const int wn   = w & 1;                        // 0..1 -> tile cols {4wn..4wn+3}
  const int n0   = blockIdx.x * 128;
  const int m0   = blockIdx.y * 128;

  const __bf16* arow0 =
      xb + (size_t)(m0 + (wm * 2 + 0) * 16 + ln) * D + 16 * half;
  const __bf16* arow1 =
      xb + (size_t)(m0 + (wm * 2 + 1) * 16 + ln) * D + 16 * half;
  const __bf16* brow[4];
  #pragma unroll
  for (int nt = 0; nt < 4; ++nt)
    brow[nt] = Wt + (size_t)(n0 + (wn * 4 + nt) * 16 + ln) * D + 16 * half;

  v8f zero = {};
  v8f acc[2][4];
  #pragma unroll
  for (int mt = 0; mt < 2; ++mt)
    #pragma unroll
    for (int nt = 0; nt < 4; ++nt) acc[mt][nt] = zero;

  for (int k0 = 0; k0 < D; k0 += 32) {
    v16bf a0 = cat16(*(const v8bf*)(arow0 + k0), *(const v8bf*)(arow0 + k0 + 8));
    v16bf a1 = cat16(*(const v8bf*)(arow1 + k0), *(const v8bf*)(arow1 + k0 + 8));
    #pragma unroll
    for (int nt = 0; nt < 4; ++nt) {
      v16bf bf = *(const v16bf*)(brow[nt] + k0);   // contiguous 32B
      acc[0][nt] = WMMA_BF16(a0, bf, acc[0][nt]);
      acc[1][nt] = WMMA_BF16(a1, bf, acc[1][nt]);
    }
    if (k0 + 32 < D) __builtin_prefetch(arow0 + k0 + 32, 0, 1);
  }

  // ---- epilogue: bias + bf16 store (optionally transposed per batch)
  #pragma unroll
  for (int nt = 0; nt < 4; ++nt) {
    const int n = n0 + (wn * 4 + nt) * 16 + ln;
    const float bv = bias[n];
    #pragma unroll
    for (int mt = 0; mt < 2; ++mt) {
      #pragma unroll
      for (int j = 0; j < 8; ++j) {
        const int m = m0 + (wm * 2 + mt) * 16 + 8 * half + j;
        const __bf16 val = (__bf16)(acc[mt][nt][j] + bv);
        if (!transpose_out) {
          out[(size_t)m * D + n] = val;
        } else {
          // Vt layout: [batch][D][S]; m = batch*S + s  (S = 2048)
          const int bb = m >> 11;
          const int s  = m & (SEQ - 1);
          out[((size_t)bb * D + n) * SEQ + s] = val;
        }
      }
    }
  }
}

// -----------------------------------------------------------------------------
// Fused attention: one block = 8 waves handles one (batch, 16-query-row) tile.
// Phase 1: wave w computes scores for key columns [w*256, w*256+256) via WMMA
//          (K-operand = contiguous 32B loads from row-major K).
// Softmax: shfl_xor row reductions within 16-lane halves + cross-wave LDS.
// P (16x2048 bf16, 64KB) staged in LDS; Phase 3: wave w computes output
// columns [w*128, w*128+128) = P @ V via WMMA; V is pre-transposed (Vt[D][S])
// so B-fragments are contiguous 32B loads. fp32 store.
// -----------------------------------------------------------------------------
__global__ __launch_bounds__(256) void attn_kernel(
    const __bf16* __restrict__ Qb, const __bf16* __restrict__ Kb,
    const __bf16* __restrict__ Vt, float* __restrict__ out)
{
  constexpr int S = 2048;
  constexpr int D = 1024;
  constexpr float scale = 0.03125f;              // 1/sqrt(1024)

  __shared__ __bf16 Pbuf[16 * S];                // 64 KB
  float* scratch = (float*)Pbuf;                 // aliased, barrier-ordered

  const int tid  = threadIdx.x;
  const int w    = tid >> 5;
  const int lane = tid & 31;
  const int half = lane >> 4;
  const int ln   = lane & 15;
  const int q0   = blockIdx.x * 16;
  const int b    = blockIdx.y;

  const __bf16* Qrow  = Qb + ((size_t)b * S + q0 + ln) * D + 16 * half;
  const __bf16* Kbase = Kb + (size_t)b * S * D;
  const __bf16* Vtb   = Vt + (size_t)b * D * S;

  // ---------------- Phase 1: raw scores, 16 key tiles per wave --------------
  v8f zero = {};
  v8f acc[16];
  #pragma unroll
  for (int t = 0; t < 16; ++t) acc[t] = zero;

  for (int d0 = 0; d0 < D; d0 += 32) {
    v8bf qlo = *(const v8bf*)(Qrow + d0);
    v8bf qhi = *(const v8bf*)(Qrow + d0 + 8);
    v16bf a = cat16(qlo, qhi);
    #pragma unroll
    for (int t = 0; t < 16; ++t) {
      const __bf16* kp =
          Kbase + (size_t)(w * 256 + t * 16 + ln) * D + d0 + 16 * half;
      v16bf bfr = *(const v16bf*)kp;             // contiguous 32B
      acc[t] = WMMA_BF16(a, bfr, acc[t]);
    }
  }

  // ---------------- Softmax: row max ----------------------------------------
  float rmax[8];
  #pragma unroll
  for (int j = 0; j < 8; ++j) rmax[j] = -3.0e38f;
  #pragma unroll
  for (int t = 0; t < 16; ++t) {
    #pragma unroll
    for (int j = 0; j < 8; ++j) {
      float s = acc[t][j] * scale;
      acc[t][j] = s;
      rmax[j] = fmaxf(rmax[j], s);
    }
  }
  #pragma unroll
  for (int m = 1; m < 16; m <<= 1) {
    #pragma unroll
    for (int j = 0; j < 8; ++j)
      rmax[j] = fmaxf(rmax[j], __shfl_xor(rmax[j], m, 32));
  }
  if (ln == 0) {
    #pragma unroll
    for (int j = 0; j < 8; ++j) scratch[w * 16 + 8 * half + j] = rmax[j];
  }
  __syncthreads();
  #pragma unroll
  for (int j = 0; j < 8; ++j) {
    const int row = 8 * half + j;
    float m = -3.0e38f;
    #pragma unroll
    for (int w2 = 0; w2 < 8; ++w2) m = fmaxf(m, scratch[w2 * 16 + row]);
    rmax[j] = m;
  }
  __syncthreads();

  // ---------------- Softmax: exp + row sum ----------------------------------
  float rsum[8];
  #pragma unroll
  for (int j = 0; j < 8; ++j) rsum[j] = 0.0f;
  #pragma unroll
  for (int t = 0; t < 16; ++t) {
    #pragma unroll
    for (int j = 0; j < 8; ++j) {
      float p = __expf(acc[t][j] - rmax[j]);
      acc[t][j] = p;
      rsum[j] += p;
    }
  }
  #pragma unroll
  for (int m = 1; m < 16; m <<= 1) {
    #pragma unroll
    for (int j = 0; j < 8; ++j) rsum[j] += __shfl_xor(rsum[j], m, 32);
  }
  if (ln == 0) {
    #pragma unroll
    for (int j = 0; j < 8; ++j) scratch[w * 16 + 8 * half + j] = rsum[j];
  }
  __syncthreads();
  #pragma unroll
  for (int j = 0; j < 8; ++j) {
    const int row = 8 * half + j;
    float s = 0.0f;
    #pragma unroll
    for (int w2 = 0; w2 < 8; ++w2) s += scratch[w2 * 16 + row];
    rsum[j] = 1.0f / s;
  }
  __syncthreads();                               // scratch reads done before P writes

  // ---------------- Write normalized P (bf16) to LDS ------------------------
  #pragma unroll
  for (int t = 0; t < 16; ++t) {
    #pragma unroll
    for (int j = 0; j < 8; ++j) {
      Pbuf[(size_t)(8 * half + j) * S + w * 256 + t * 16 + ln] =
          (__bf16)(acc[t][j] * rsum[j]);
    }
  }
  __syncthreads();

  // ---------------- Phase 3: O = P @ V, 8 column tiles per wave -------------
  v8f o[8];
  #pragma unroll
  for (int t = 0; t < 8; ++t) o[t] = zero;

  const __bf16* Prow = Pbuf + (size_t)ln * S + 16 * half;
  for (int j0 = 0; j0 < S; j0 += 32) {
    v8bf plo = *(const v8bf*)(Prow + j0);
    v8bf phi = *(const v8bf*)(Prow + j0 + 8);
    v16bf a = cat16(plo, phi);
    #pragma unroll
    for (int t = 0; t < 8; ++t) {
      // B[k][n] = V[j0+k][n] = Vt[n][j0+k]: lane holds col n, k contiguous
      const __bf16* vp =
          Vtb + (size_t)(w * 128 + t * 16 + ln) * S + j0 + 16 * half;
      v16bf bfr = *(const v16bf*)vp;             // contiguous 32B
      o[t] = WMMA_BF16(a, bfr, o[t]);
    }
  }

  #pragma unroll
  for (int t = 0; t < 8; ++t) {
    #pragma unroll
    for (int j = 0; j < 8; ++j) {
      out[((size_t)b * S + q0 + 8 * half + j) * D + w * 128 + t * 16 + ln] =
          o[t][j];
    }
  }
}

// -----------------------------------------------------------------------------
extern "C" void kernel_launch(void* const* d_in, const int* in_sizes, int n_in,
                              void* d_out, int out_size, void* d_ws, size_t ws_size,
                              hipStream_t stream) {
  (void)in_sizes; (void)n_in; (void)out_size; (void)ws_size;
  const float* x  = (const float*)d_in[0];
  const float* Wq = (const float*)d_in[1];
  const float* bq = (const float*)d_in[2];
  const float* Wk = (const float*)d_in[3];
  const float* bk = (const float*)d_in[4];
  const float* Wv = (const float*)d_in[5];
  const float* bv = (const float*)d_in[6];
  float* out = (float*)d_out;

  constexpr int B = 8, S = 2048, D = 1024;
  constexpr int M = B * S;

  __bf16* Qb = (__bf16*)d_ws;                    // 32 MB, [M][D]
  __bf16* Kb = Qb + (size_t)M * D;               // 32 MB, [M][D]
  __bf16* Vt = Kb + (size_t)M * D;               // 32 MB, [B][D][S] (transposed)
  __bf16* xb = Vt + (size_t)M * D;               // 32 MB, [M][D] bf16 copy of x
  __bf16* Wt = xb + (size_t)M * D;               //  2 MB, [D][D] n-major (reused)

  // one-time fp32 -> bf16 conversion of x
  cvt_bf16_kernel<<<(int)(((size_t)M * D) / 8 / 256), 256, 0, stream>>>(x, xb);

  dim3 gtr(D / 64, D / 64);                      // (16, 16)
  dim3 gproj(D / 128, M / 128);                  // (8, 128)

  wtrans_kernel<<<gtr, 256, 0, stream>>>(Wq, Wt, D);
  qkv_gemm_kernel<<<gproj, 256, 0, stream>>>(xb, Wt, bq, Qb, D, 0);

  wtrans_kernel<<<gtr, 256, 0, stream>>>(Wk, Wt, D);
  qkv_gemm_kernel<<<gproj, 256, 0, stream>>>(xb, Wt, bk, Kb, D, 0);

  wtrans_kernel<<<gtr, 256, 0, stream>>>(Wv, Wt, D);
  qkv_gemm_kernel<<<gproj, 256, 0, stream>>>(xb, Wt, bv, Vt, D, 1);

  attn_kernel<<<dim3(S / 16, B), 256, 0, stream>>>(Qb, Kb, Vt, out);
}
